// BioSeqMixerModel_37606733644547
// MI455X (gfx1250) — matrix operand order repaired and checked by reading, not compiled, not run on hardware
//
#include <hip/hip_runtime.h>
#include <hip/hip_bf16.h>

// ---------------------------------------------------------------------------
// Mamba forward for MI455X (gfx1250, wave32).
// GEMMs: v_wmma_f32_16x16x32_bf16 with 2x2 register tiling per wave
// (4 WMMAs per 8 LDS fragment loads). bf16 operands pre-converted/pre-padded
// so the inner loop is pure b128 movement + WMMA. Scan/LN/conv: fp32 VALU.
// ---------------------------------------------------------------------------

#define Bsz     4
#define Lsz     1024
#define BL      (Bsz * Lsz)            // 4096
#define DM      768
#define DI      1536
#define DSTATE  16
#define DCONV   4
#define DTRANK  48
#define DTPAD   64                     // dt_rank padded to K multiple of 32
#define NLAYER  4
#define DBCW    (DTRANK + 2 * DSTATE)  // 80
#define ASTR    56                     // LDS row stride (ushort): 112B = 28-bank
                                       // rotation -> conflict-free frag loads

typedef unsigned short v8us  __attribute__((ext_vector_type(8)));
typedef unsigned short v16us __attribute__((ext_vector_type(16)));
typedef __bf16         v16bf __attribute__((ext_vector_type(16)));
typedef float          v8f   __attribute__((ext_vector_type(8)));

__device__ __forceinline__ unsigned short f2bf(float f) {
    unsigned u = __builtin_bit_cast(unsigned, f);
    u += 0x7FFFu + ((u >> 16) & 1u);           // round-to-nearest-even
    return (unsigned short)(u >> 16);
}
__device__ __forceinline__ unsigned pack_bf2(float lo, float hi) {
    return (unsigned)f2bf(lo) | ((unsigned)f2bf(hi) << 16);
}
__device__ __forceinline__ float silu_f(float x) {
    return x / (1.f + __expf(-x));
}
__device__ __forceinline__ v8f wmma_bf16(v16us a, v16us b, v8f c) {
    return __builtin_amdgcn_wmma_f32_16x16x32_bf16(
        false, __builtin_bit_cast(v16bf, a),
        false, __builtin_bit_cast(v16bf, b),
        (short)0, c, false, false);
}

// ---------------------------------------------------------------------------
// C[M,N] = A[M,K] * W[N,K]^T.  A, W already bf16.  K multiple of 32, M % 128 == 0.
// MODE 0: store   MODE 1: softplus(acc + bias[n])   MODE 2: C += acc
// BN: N needs bounds checking (only x_proj, N=80).
// Block: 256 threads = 8 waves; block tile 128(M) x 64(N); wave tile 32x32
// (2x2 WMMA tiles); K-step 32.
// ---------------------------------------------------------------------------
template<int MODE, bool BN>
__global__ __launch_bounds__(256) void gemm_bf16_wmma(
    const unsigned short* __restrict__ A, const unsigned short* __restrict__ W,
    const float* __restrict__ bias, float* __restrict__ C,
    int M, int N, int K, int lda, int ldw, int ldc)
{
    __shared__ unsigned short As[128][ASTR];   // [m][k], 32 cols used
    __shared__ unsigned short Bs[64][ASTR];    // [n][k], 32 cols used

    const int tid  = threadIdx.x;
    const int lane = tid & 31;
    const int wid  = tid >> 5;
    const int wm   = wid & 3;               // wave M rows [wm*32, +32)
    const int wn   = wid >> 2;              // wave N cols [wn*32, +32)
    const int half = lane >> 4;
    const int l15  = lane & 15;

    const int blockM = blockIdx.y * 128;
    const int blockN = blockIdx.x * 64;

    // cooperative staging coordinates
    const int arow = tid >> 1;              // 0..127, 32B per thread
    const int acol = (tid & 1) * 16;
    const int brow = tid >> 2;              // 0..63, 16B per thread
    const int bcol = (tid & 3) * 8;

    const unsigned short* aSrc = A + (size_t)(blockM + arow) * lda + acol;
    const bool bok = !BN || (blockN + brow) < N;
    const unsigned short* bSrc = W + (size_t)(blockN + brow) * ldw + bcol;

    v8f acc00 = {}, acc01 = {}, acc10 = {}, acc11 = {};
    const int nk = K >> 5;

    for (int kt = 0; kt < nk; ++kt) {
        const int kbase = kt << 5;
        __syncthreads();
        // stage A tile: 2 x b128 per thread
        const uint4 a0 = *(const uint4*)(aSrc + kbase);
        const uint4 a1 = *(const uint4*)(aSrc + kbase + 8);
        *(uint4*)&As[arow][acol]     = a0;
        *(uint4*)&As[arow][acol + 8] = a1;
        // stage B tile: 1 x b128 per thread
        uint4 bv; bv.x = 0u; bv.y = 0u; bv.z = 0u; bv.w = 0u;
        if (bok) bv = *(const uint4*)(bSrc + kbase);
        *(uint4*)&Bs[brow][bcol] = bv;
        // prefetch next K tile (global_prefetch_b8)
        if (kt + 1 < nk) {
            __builtin_prefetch(aSrc + kbase + 32, 0, 1);
            if (bok) __builtin_prefetch(bSrc + kbase + 32, 0, 1);
        }
        __syncthreads();

        // A fragments (contiguous K: half selects K 0-15 / 16-31): rows wm*32 + {0,16} + l15
        const int m0 = wm * 32 + l15;
        v8us a00 = *(const v8us*)&As[m0][half * 16];
        v8us a01 = *(const v8us*)&As[m0][half * 16 + 8];
        v16us au0 = __builtin_shufflevector(a00, a01,
            0,1,2,3,4,5,6,7,8,9,10,11,12,13,14,15);
        v8us a10 = *(const v8us*)&As[m0 + 16][half * 16];
        v8us a11 = *(const v8us*)&As[m0 + 16][half * 16 + 8];
        v16us au1 = __builtin_shufflevector(a10, a11,
            0,1,2,3,4,5,6,7,8,9,10,11,12,13,14,15);
        // B fragments: cols wn*32 + {0,16} + l15; half selects K 0-15 / 16-31
        const int n0 = wn * 32 + l15;
        v16us bu0 = *(const v16us*)&Bs[n0][half * 16];
        v16us bu1 = *(const v16us*)&Bs[n0 + 16][half * 16];

        acc00 = wmma_bf16(au0, bu0, acc00);
        acc01 = wmma_bf16(au0, bu1, acc01);
        acc10 = wmma_bf16(au1, bu0, acc10);
        acc11 = wmma_bf16(au1, bu1, acc11);
    }

    // epilogue: VGPR r -> row (r + 8*half), col (lane&15)
    const int mgBase = blockM + wm * 32 + half * 8;
    const int ngA = blockN + wn * 32 + l15;
    const int ngB = ngA + 16;

    auto emit = [&](const v8f& accv, int mrow, int ng) {
        if (BN && ng >= N) return;
        const float bval = (MODE == 1) ? bias[ng] : 0.f;
        #pragma unroll
        for (int r = 0; r < 8; ++r) {
            const size_t off = (size_t)(mrow + r) * ldc + ng;
            float v = accv[r];
            if (MODE == 1) {
                v += bval;
                v = (v > 20.f) ? v : log1pf(__expf(v));
                C[off] = v;
            } else if (MODE == 2) {
                C[off] += v;
            } else {
                C[off] = v;
            }
        }
    };
    emit(acc00, mgBase,      ngA);
    emit(acc01, mgBase,      ngB);
    emit(acc10, mgBase + 16, ngA);
    emit(acc11, mgBase + 16, ngB);
}

// ---------------------------------------------------------------------------
// f32 -> bf16, 8 elements per thread, fully vectorized. n8 = count/8.
__global__ __launch_bounds__(256) void cvt_bf16_kernel(
    const float* __restrict__ in, unsigned short* __restrict__ out, int n8)
{
    const int i = blockIdx.x * 256 + threadIdx.x;
    if (i >= n8) return;
    const float4 f0 = ((const float4*)in)[i * 2];
    const float4 f1 = ((const float4*)in)[i * 2 + 1];
    uint4 o;
    o.x = pack_bf2(f0.x, f0.y);
    o.y = pack_bf2(f0.z, f0.w);
    o.z = pack_bf2(f1.x, f1.y);
    o.w = pack_bf2(f1.z, f1.w);
    ((uint4*)out)[i] = o;
}

// f32[rows x ldin] (first kin cols) -> bf16[rows x kout], zero-padded tail.
__global__ __launch_bounds__(256) void cvt_pad_kernel(
    const float* __restrict__ in, unsigned short* __restrict__ out,
    int rows, int ldin, int kin, int kout)
{
    const int idx = blockIdx.x * 256 + threadIdx.x;
    if (idx >= rows * kout) return;
    const int r = idx / kout, k = idx % kout;
    out[idx] = (k < kin) ? f2bf(in[(size_t)r * ldin + k]) : (unsigned short)0;
}

// ---------------------------------------------------------------------------
__global__ __launch_bounds__(256) void embed_kernel(
    const int* __restrict__ ids, const float* __restrict__ embed,
    float* __restrict__ resid)
{
    const int idx = blockIdx.x * 256 + threadIdx.x;     // BL*DM threads
    const int bl = idx / DM, d = idx % DM;
    resid[idx] = embed[(size_t)ids[bl] * DM + d];
}

// ---------------------------------------------------------------------------
// LayerNorm; BF16OUT selects fused bf16 emission (feeds in_proj GEMM).
template<bool BF16OUT>
__global__ __launch_bounds__(256) void ln_kernel(
    const float* __restrict__ x, const float* __restrict__ w,
    const float* __restrict__ b, float* __restrict__ outf,
    unsigned short* __restrict__ outb)
{
    __shared__ float red[256];
    const int row = blockIdx.x;
    const float* xr = x + (size_t)row * DM;

    float s = 0.f;
    for (int d = threadIdx.x; d < DM; d += 256) s += xr[d];
    red[threadIdx.x] = s; __syncthreads();
    for (int o = 128; o > 0; o >>= 1) {
        if (threadIdx.x < o) red[threadIdx.x] += red[threadIdx.x + o];
        __syncthreads();
    }
    const float mu = red[0] * (1.f / DM);
    __syncthreads();

    float vs = 0.f;
    for (int d = threadIdx.x; d < DM; d += 256) { float t = xr[d] - mu; vs += t * t; }
    red[threadIdx.x] = vs; __syncthreads();
    for (int o = 128; o > 0; o >>= 1) {
        if (threadIdx.x < o) red[threadIdx.x] += red[threadIdx.x + o];
        __syncthreads();
    }
    const float rstd = rsqrtf(red[0] * (1.f / DM) + 1e-5f);

    for (int d = threadIdx.x; d < DM; d += 256) {
        const float v = (xr[d] - mu) * rstd * w[d] + b[d];
        if (BF16OUT) outb[(size_t)row * DM + d] = f2bf(v);
        else         outf[(size_t)row * DM + d] = v;
    }
}

// ---------------------------------------------------------------------------
// depthwise causal conv (4 taps) + SiLU; emits both f32 (scan) and bf16 (GEMM A)
__global__ __launch_bounds__(256) void conv_silu_kernel(
    const float* __restrict__ xz, const float* __restrict__ cw,
    const float* __restrict__ cb, float* __restrict__ xc,
    unsigned short* __restrict__ xcb)
{
    const int idx = blockIdx.x * 256 + threadIdx.x;     // BL*DI threads
    const int d = idx % DI;
    const int bl = idx / DI;
    const int l = bl & (Lsz - 1);
    float acc = cb[d];
    #pragma unroll
    for (int k = 0; k < DCONV; ++k) {
        const int ls = l + k - (DCONV - 1);
        if (ls >= 0)
            acc += cw[d * DCONV + k] * xz[(size_t)(bl + k - (DCONV - 1)) * (2 * DI) + d];
    }
    const float v = silu_f(acc);
    xc[idx]  = v;
    xcb[idx] = f2bf(v);
}

// ---------------------------------------------------------------------------
// selective scan: one thread per (b, d) channel; 16-state in registers.
// Fused epilogue: y = (y + x*D) * silu(z), emitted as bf16 (feeds out_proj).
__global__ __launch_bounds__(256) void scan_kernel(
    const float* __restrict__ xc, const float* __restrict__ dt,
    const float* __restrict__ dbc, const float* __restrict__ xz,
    const float* __restrict__ A_log, const float* __restrict__ Dp,
    unsigned short* __restrict__ yfb)
{
    const int t = blockIdx.x * 256 + threadIdx.x;       // Bsz*DI threads
    const int b = t / DI;
    const int d = t % DI;

    float A2[DSTATE];
    #pragma unroll
    for (int n = 0; n < DSTATE; ++n)
        A2[n] = -__expf(A_log[(size_t)d * DSTATE + n]) * 1.44269504f; // A * log2(e)

    float st[DSTATE];
    #pragma unroll
    for (int n = 0; n < DSTATE; ++n) st[n] = 0.f;

    const float dcoef = Dp[d];

    for (int l = 0; l < Lsz; ++l) {
        const int bl = b * Lsz + l;
        const float x   = xc[(size_t)bl * DI + d];
        const float dtv = dt[(size_t)bl * DI + d];
        const float* Brow = dbc + (size_t)bl * DBCW + DTRANK;
        const float* Crow = Brow + DSTATE;
        const float dtx = dtv * x;
        float y = 0.f;
        #pragma unroll
        for (int n = 0; n < DSTATE; ++n) {
            const float dA = exp2f(dtv * A2[n]);
            st[n] = st[n] * dA + dtx * Brow[n];
            y += st[n] * Crow[n];
        }
        const float z = xz[(size_t)bl * (2 * DI) + DI + d];
        yfb[(size_t)bl * DI + d] = f2bf((y + x * dcoef) * silu_f(z));
    }
}

// ---------------------------------------------------------------------------
extern "C" void kernel_launch(void* const* d_in, const int* in_sizes, int n_in,
                              void* d_out, int out_size, void* d_ws, size_t ws_size,
                              hipStream_t stream)
{
    const int*   ids        = (const int*)  d_in[0];
    const float* embed      = (const float*)d_in[1];
    const float* in_proj_w  = (const float*)d_in[2];
    const float* conv_w     = (const float*)d_in[3];
    const float* conv_b     = (const float*)d_in[4];
    const float* x_proj_w   = (const float*)d_in[5];
    const float* dt_proj_w  = (const float*)d_in[6];
    const float* dt_proj_b  = (const float*)d_in[7];
    const float* A_log      = (const float*)d_in[8];
    const float* Dp         = (const float*)d_in[9];
    const float* out_proj_w = (const float*)d_in[10];
    const float* norm_w     = (const float*)d_in[11];
    const float* norm_b     = (const float*)d_in[12];
    const float* norm_f_w   = (const float*)d_in[13];
    const float* norm_f_b   = (const float*)d_in[14];

    char* wsp = (char*)d_ws;
    auto carveF = [&](size_t n) { float* p = (float*)wsp; wsp += n * 4; return p; };
    auto carveH = [&](size_t n) { unsigned short* p = (unsigned short*)wsp; wsp += n * 2; return p; };

    // fp32 activations
    float* resid = carveF((size_t)BL * DM);
    float* xz    = carveF((size_t)BL * 2 * DI);
    float* xc    = carveF((size_t)BL * DI);
    float* dbc   = carveF((size_t)BL * DBCW);
    float* dtb   = carveF((size_t)BL * DI);
    // bf16 GEMM operands (activations)
    unsigned short* h_bf    = carveH((size_t)BL * DM);
    unsigned short* xc_bf   = carveH((size_t)BL * DI);
    unsigned short* dtlo_bf = carveH((size_t)BL * DTPAD);
    unsigned short* yf_bf   = carveH((size_t)BL * DI);
    // bf16 weights (all layers, converted once per call)
    unsigned short* wi_bf  = carveH((size_t)NLAYER * 2 * DI * DM);
    unsigned short* wx_bf  = carveH((size_t)NLAYER * DBCW * DI);
    unsigned short* wdt_bf = carveH((size_t)NLAYER * DI * DTPAD);
    unsigned short* wo_bf  = carveH((size_t)NLAYER * DM * DI);

    // ---- one-time weight conversion ----
    {
        const int n1 = NLAYER * 2 * DI * DM / 8;
        cvt_bf16_kernel<<<(n1 + 255) / 256, 256, 0, stream>>>(in_proj_w, wi_bf, n1);
        const int n2 = NLAYER * DBCW * DI / 8;
        cvt_bf16_kernel<<<(n2 + 255) / 256, 256, 0, stream>>>(x_proj_w, wx_bf, n2);
        const int n3 = NLAYER * DM * DI / 8;
        cvt_bf16_kernel<<<(n3 + 255) / 256, 256, 0, stream>>>(out_proj_w, wo_bf, n3);
        const int rows = NLAYER * DI;                       // pad K 48 -> 64
        cvt_pad_kernel<<<(rows * DTPAD + 255) / 256, 256, 0, stream>>>(
            dt_proj_w, wdt_bf, rows, DTRANK, DTRANK, DTPAD);
    }

    embed_kernel<<<(BL * DM) / 256, 256, 0, stream>>>(ids, embed, resid);

    for (int i = 0; i < NLAYER; ++i) {
        // h = LN(resid) -> bf16
        ln_kernel<true><<<BL, 256, 0, stream>>>(
            resid, norm_w + i * DM, norm_b + i * DM, nullptr, h_bf);

        // xz = h @ Win^T : (4096 x 3072 x 768)
        dim3 g1((2 * DI) / 64, BL / 128);
        gemm_bf16_wmma<0, false><<<g1, 256, 0, stream>>>(
            h_bf, wi_bf + (size_t)i * 2 * DI * DM, nullptr, xz,
            BL, 2 * DI, DM, DM, DM, 2 * DI);

        conv_silu_kernel<<<(BL * DI) / 256, 256, 0, stream>>>(
            xz, conv_w + (size_t)i * DI * DCONV, conv_b + (size_t)i * DI, xc, xc_bf);

        // dbc = xc @ Wx^T : (4096 x 80 x 1536), N needs bounds
        dim3 g2((DBCW + 63) / 64, BL / 128);
        gemm_bf16_wmma<0, true><<<g2, 256, 0, stream>>>(
            xc_bf, wx_bf + (size_t)i * DBCW * DI, nullptr, dbc,
            BL, DBCW, DI, DI, DI, DBCW);

        // dt_lo = dbc[:, :48] -> bf16 padded to 64 cols
        cvt_pad_kernel<<<(BL * DTPAD) / 256, 256, 0, stream>>>(
            dbc, dtlo_bf, BL, DBCW, DTRANK, DTPAD);

        // dt = softplus(dt_lo @ Wdt^T + bdt) : (4096 x 1536 x 64)
        dim3 g3(DI / 64, BL / 128);
        gemm_bf16_wmma<1, false><<<g3, 256, 0, stream>>>(
            dtlo_bf, wdt_bf + (size_t)i * DI * DTPAD, dt_proj_b + (size_t)i * DI, dtb,
            BL, DI, DTPAD, DTPAD, DTPAD, DI);

        scan_kernel<<<(Bsz * DI) / 256, 256, 0, stream>>>(
            xc, dtb, dbc, xz,
            A_log + (size_t)i * DI * DSTATE, Dp + (size_t)i * DI, yf_bf);

        // resid += y @ Wout^T : (4096 x 768 x 1536)
        dim3 g4(DM / 64, BL / 128);
        gemm_bf16_wmma<2, false><<<g4, 256, 0, stream>>>(
            yf_bf, wo_bf + (size_t)i * DM * DI, nullptr, resid,
            BL, DM, DI, DI, DI, DM);
    }

    ln_kernel<false><<<BL, 256, 0, stream>>>(
        resid, norm_f_w, norm_f_b, (float*)d_out, nullptr);
}
